// UltraOptimizedMoE_11390253269261
// MI455X (gfx1250) — matrix operand order repaired and checked by reading, not compiled, not run on hardware
//
#include <hip/hip_runtime.h>
#include <hip/hip_bf16.h>
#include <math.h>

typedef __attribute__((ext_vector_type(8)))  float   v8f;
typedef __attribute__((ext_vector_type(8)))  __bf16  v8bf;
typedef __attribute__((ext_vector_type(16))) __bf16  v16bf;

#define EPSV 1e-5f
#define LDS_PITCH 40   // ushorts; 80B row pitch: 16B-aligned chunks, conflict-free

// native conversions: backend emits hardware cvt path for gfx1250
__device__ __forceinline__ unsigned short f2bf(float f) {
  return __builtin_bit_cast(unsigned short, (__bf16)f);
}
__device__ __forceinline__ float bf2f(unsigned short h) {
  return (float)__builtin_bit_cast(__bf16, h);
}
__device__ __forceinline__ float silu(float v) { return v / (1.f + __expf(-v)); }

// build ISA-layout fragment from two aligned 16B LDS chunks
__device__ __forceinline__ v16bf frag16(const unsigned short* p0,
                                        const unsigned short* p1) {
  v8bf a0 = *(const v8bf*)p0;
  v8bf a1 = *(const v8bf*)p1;
  return __builtin_shufflevector(a0, a1, 0, 1, 2, 3, 4, 5, 6, 7,
                                 8, 9, 10, 11, 12, 13, 14, 15);
}

// ---------------------------------------------------------------------------
// f32 -> bf16 pre-conversion (vectorized, 4 elements / thread)
// ---------------------------------------------------------------------------
__global__ __launch_bounds__(256) void cvt_bf16(const float* __restrict__ src,
                                                unsigned short* __restrict__ dst,
                                                int n4) {
  int i = blockIdx.x * 256 + threadIdx.x;
  if (i >= n4) return;
  float4 f = ((const float4*)src)[i];
  uint2 pk;
  pk.x = (unsigned)f2bf(f.x) | ((unsigned)f2bf(f.y) << 16);
  pk.y = (unsigned)f2bf(f.z) | ((unsigned)f2bf(f.w) << 16);
  ((uint2*)dst)[i] = pk;
}

// ---------------------------------------------------------------------------
// Router stage 1: 8x8 average pool  x[32,256,64,64] -> xd[32,256,8,8]
// ---------------------------------------------------------------------------
__global__ __launch_bounds__(256) void r_pool(const float* __restrict__ x,
                                              float* __restrict__ xd) {
  int idx = blockIdx.x * 256 + threadIdx.x;          // < 524288
  int p = idx & 63;
  int py = p >> 3, px = p & 7;
  const float* src = x + ((size_t)(idx >> 6)) * 4096 + (py * 8) * 64 + px * 8;
  float s = 0.f;
  #pragma unroll
  for (int iy = 0; iy < 8; ++iy)
    #pragma unroll
    for (int ix = 0; ix < 8; ++ix)
      s += src[iy * 64 + ix];
  xd[idx] = s * (1.f / 64.f);
}

// ---------------------------------------------------------------------------
// Router stage 2: depthwise 3x3 SAME + GroupNorm(8) + SiLU  (one block/image)
// ---------------------------------------------------------------------------
__global__ __launch_bounds__(256) void r_dwgn(const float* __restrict__ xd,
                                              const float* __restrict__ dw,
                                              const float* __restrict__ g1,
                                              const float* __restrict__ b1,
                                              float* __restrict__ h) {
  extern __shared__ float smem[];
  float* conv = smem;             // 16384
  float* red  = smem + 16384;     // 256
  float* red2 = red + 256;        // 256
  int b = blockIdx.x, c = threadIdx.x;
  const float* src = xd + ((size_t)b * 256 + c) * 64;
  const float* k9  = dw + c * 9;
  float s = 0.f, s2 = 0.f;
  for (int p = 0; p < 64; ++p) {
    int y = p >> 3, xq = p & 7;
    float a = 0.f;
    #pragma unroll
    for (int ky = -1; ky <= 1; ++ky)
      #pragma unroll
      for (int kx = -1; kx <= 1; ++kx) {
        int yy = y + ky, xx = xq + kx;
        if (yy >= 0 && yy < 8 && xx >= 0 && xx < 8)
          a += k9[(ky + 1) * 3 + (kx + 1)] * src[yy * 8 + xx];
      }
    conv[c * 64 + p] = a;
    s += a; s2 += a * a;
  }
  red[c] = s; red2[c] = s2;
  __syncthreads();
  int g = c >> 5;                                   // 32 channels / group
  float gs = 0.f, gs2 = 0.f;
  for (int j = 0; j < 32; ++j) { gs += red[g * 32 + j]; gs2 += red2[g * 32 + j]; }
  float mean = gs * (1.f / 2048.f);
  float rs = rsqrtf(gs2 * (1.f / 2048.f) - mean * mean + EPSV);
  float gg = g1[c], bb = b1[c];
  for (int p = 0; p < 64; ++p) {
    float v = (conv[c * 64 + p] - mean) * rs * gg + bb;
    h[((size_t)b * 256 + c) * 64 + p] = silu(v);
  }
}

// ---------------------------------------------------------------------------
// Router stage 3: pw1 + GN(4) + SiLU + pw2 + softmax + pool + top2 + threshold
// ---------------------------------------------------------------------------
__global__ __launch_bounds__(64) void r_route(const float* __restrict__ h,
                                              const float* __restrict__ pw1,
                                              const float* __restrict__ g2,
                                              const float* __restrict__ b2,
                                              const float* __restrict__ pw2,
                                              const float* __restrict__ pb2,
                                              int* __restrict__ eidx,
                                              float* __restrict__ wslot) {
  extern __shared__ float smem[];
  float* hs     = smem;                 // 16384
  float* part   = smem + 16384;         // 256
  float* part2  = part + 256;           // 256
  float* pool   = part2 + 256;          // 512
  float* pooled = pool + 512;           // 8
  int b = blockIdx.x, p = threadIdx.x;
  for (int i = 0; i < 256; ++i) hs[i * 64 + p] = h[((size_t)b * 256 + i) * 64 + p];
  __syncthreads();
  float h2[16];
  #pragma unroll
  for (int r = 0; r < 16; ++r) {
    float a = 0.f;
    for (int cc = 0; cc < 256; ++cc) a += pw1[r * 256 + cc] * hs[cc * 64 + p];
    h2[r] = a;
  }
  #pragma unroll
  for (int g = 0; g < 4; ++g) {
    float s = 0.f, s2 = 0.f;
    #pragma unroll
    for (int j = 0; j < 4; ++j) { float v = h2[4 * g + j]; s += v; s2 += v * v; }
    part[g * 64 + p] = s; part2[g * 64 + p] = s2;
  }
  __syncthreads();
  float mean[4], rs[4];
  #pragma unroll
  for (int g = 0; g < 4; ++g) {
    float gs = 0.f, gs2 = 0.f;
    for (int j = 0; j < 64; ++j) { gs += part[g * 64 + j]; gs2 += part2[g * 64 + j]; }
    mean[g] = gs * (1.f / 256.f);
    rs[g] = rsqrtf(gs2 * (1.f / 256.f) - mean[g] * mean[g] + EPSV);
  }
  #pragma unroll
  for (int r = 0; r < 16; ++r) {
    int g = r >> 2;
    h2[r] = silu((h2[r] - mean[g]) * rs[g] * g2[r] + b2[r]);
  }
  float z[8]; float zmax = -1e30f;
  #pragma unroll
  for (int e = 0; e < 8; ++e) {
    float a = pb2[e];
    #pragma unroll
    for (int r = 0; r < 16; ++r) a += pw2[e * 16 + r] * h2[r];
    z[e] = a; zmax = fmaxf(zmax, a);
  }
  float zs = 0.f;
  #pragma unroll
  for (int e = 0; e < 8; ++e) { z[e] = __expf(z[e] - zmax); zs += z[e]; }
  #pragma unroll
  for (int e = 0; e < 8; ++e) pool[e * 64 + p] = z[e] / zs;
  __syncthreads();
  if (p < 8) {
    float s = 0.f;
    for (int j = 0; j < 64; ++j) s += pool[p * 64 + j];
    pooled[p] = s * (1.f / 64.f);
  }
  __syncthreads();
  if (p == 0) {
    int e0 = 0; float v0 = pooled[0];
    for (int e = 1; e < 8; ++e) if (pooled[e] > v0) { v0 = pooled[e]; e0 = e; }
    int e1 = (e0 == 0) ? 1 : 0; float v1 = pooled[e1];
    for (int e = 0; e < 8; ++e)
      if (e != e0 && pooled[e] > v1) { v1 = pooled[e]; e1 = e; }
    float t = v0 + v1 + 1e-9f;
    float w0 = v0 / t, w1 = v1 / t;
    w0 = (w0 > 0.01f) ? w0 : 0.f;
    w1 = (w1 > 0.01f) ? w1 : 0.f;
    eidx[b * 2] = e0; eidx[b * 2 + 1] = e1;
    wslot[b * 2] = w0; wslot[b * 2 + 1] = w1;
  }
}

// ---------------------------------------------------------------------------
// Expert GEMM 1: y1[slot, 512, 4096] = W1[e] (512x256) @ x[b] (256x4096)
// 128x128 tile / block, 8 waves, bf16 WMMA 16x16x32, f32 accumulate.
// All operands pre-converted to bf16 -> staging is pure copy (no VALU cvt).
// LDS: A as [m][k] pitch 40, B transposed as [n][k] pitch 40 -> all fragment
// gathers are aligned ds_load_b128 pairs.
// ---------------------------------------------------------------------------
__global__ __launch_bounds__(256) void moe_gemm1(const unsigned short* __restrict__ xbf,
                                                 const unsigned short* __restrict__ w1bf,
                                                 const int* __restrict__ eidx,
                                                 unsigned short* __restrict__ y1) {
  __shared__ unsigned short lA[128 * LDS_PITCH];   // [m][k]
  __shared__ unsigned short lB[128 * LDS_PITCH];   // [n][k] (transposed)
  const int tid  = threadIdx.x;
  const int slot = blockIdx.z;
  const int mb   = blockIdx.y * 128;
  const int nb   = blockIdx.x * 128;
  const int e    = eidx[slot];
  const unsigned short* W = w1bf + (size_t)e * 512 * 256;
  const unsigned short* X = xbf  + (size_t)(slot >> 1) * 256 * 4096;
  const int wave = tid >> 5, lane = tid & 31, l = lane & 15, hi = lane >> 4;
  v8f acc[8] = {};
  for (int k0 = 0; k0 < 256; k0 += 32) {
    // A tile 128x32: b64 copies
    #pragma unroll
    for (int j = 0; j < 4; ++j) {
      int idx = tid + j * 256;
      int m = idx >> 3, k4 = (idx & 7) * 4;
      uint2 pk = *(const uint2*)&W[(size_t)(mb + m) * 256 + k0 + k4];
      *(uint2*)&lA[m * LDS_PITCH + k4] = pk;
    }
    // B tile 32x128 -> LDS [n][k]: b64 loads, transposed b16 stores
    #pragma unroll
    for (int j = 0; j < 4; ++j) {
      int idx = tid + j * 256;
      int k = idx >> 5, n4 = (idx & 31) * 4;
      uint2 raw = *(const uint2*)&X[(size_t)(k0 + k) * 4096 + nb + n4];
      lB[(n4 + 0) * LDS_PITCH + k] = (unsigned short)(raw.x & 0xFFFF);
      lB[(n4 + 1) * LDS_PITCH + k] = (unsigned short)(raw.x >> 16);
      lB[(n4 + 2) * LDS_PITCH + k] = (unsigned short)(raw.y & 0xFFFF);
      lB[(n4 + 3) * LDS_PITCH + k] = (unsigned short)(raw.y >> 16);
    }
    __syncthreads();
    // A fragment: per-lane chunks {hi*8..+7} and {16+hi*8..+7}  (ISA layout)
    const unsigned short* ar = &lA[(wave * 16 + l) * LDS_PITCH];
    v16bf a = frag16(ar + hi * 8, ar + 16 + hi * 8);
    #pragma unroll
    for (int t = 0; t < 8; ++t) {
      // B fragment: per-lane chunks {hi*8..+7} and {16+hi*8..+7} at fixed n
      const unsigned short* br = &lB[(t * 16 + l) * LDS_PITCH];
      v16bf bb = frag16(br + hi * 8, br + 16 + hi * 8);
      acc[t] = __builtin_amdgcn_wmma_f32_16x16x32_bf16(
          false, a, false, bb, (short)0, acc[t], false, false);
    }
    __syncthreads();
  }
  unsigned short* Y = y1 + (size_t)slot * 512 * 4096;
  #pragma unroll
  for (int t = 0; t < 8; ++t)
    #pragma unroll
    for (int r = 0; r < 8; ++r) {
      int m = mb + wave * 16 + hi * 8 + r;                // ISA C/D layout
      int n = nb + t * 16 + l;
      Y[(size_t)m * 4096 + n] = f2bf(acc[t][r]);
    }
}

// ---------------------------------------------------------------------------
// Per (slot, group) GroupNorm stats over bf16 activations
// ---------------------------------------------------------------------------
__global__ __launch_bounds__(256) void moe_stats(const unsigned short* __restrict__ y,
                                                 float* __restrict__ stats,
                                                 int C, int chPerGroup) {
  __shared__ float rsm[256], rsm2[256];
  int g = blockIdx.x, slot = blockIdx.y;
  const size_t base = ((size_t)slot * C + (size_t)g * chPerGroup) * 4096;
  const int count = chPerGroup * 4096;
  float s = 0.f, s2 = 0.f;
  for (int i = threadIdx.x * 4; i < count; i += 1024) {
    uint2 raw = *(const uint2*)&y[base + i];
    float v0 = bf2f((unsigned short)(raw.x & 0xFFFF));
    float v1 = bf2f((unsigned short)(raw.x >> 16));
    float v2 = bf2f((unsigned short)(raw.y & 0xFFFF));
    float v3 = bf2f((unsigned short)(raw.y >> 16));
    s += (v0 + v1) + (v2 + v3);
    s2 += (v0 * v0 + v1 * v1) + (v2 * v2 + v3 * v3);
  }
  rsm[threadIdx.x] = s; rsm2[threadIdx.x] = s2;
  __syncthreads();
  for (int off = 128; off > 0; off >>= 1) {
    if (threadIdx.x < off) {
      rsm[threadIdx.x]  += rsm[threadIdx.x + off];
      rsm2[threadIdx.x] += rsm2[threadIdx.x + off];
    }
    __syncthreads();
  }
  if (threadIdx.x == 0) {
    float mean = rsm[0] / (float)count;
    float var  = rsm2[0] / (float)count - mean * mean;
    stats[(slot * 8 + g) * 2]     = mean;
    stats[(slot * 8 + g) * 2 + 1] = rsqrtf(var + EPSV);
  }
}

// ---------------------------------------------------------------------------
// Expert GEMM 2: y2 = W2[e] (256x512) @ silu(gn1(y1))  (GN fused into B stage)
// ---------------------------------------------------------------------------
__global__ __launch_bounds__(256) void moe_gemm2(const unsigned short* __restrict__ y1,
                                                 const unsigned short* __restrict__ w2bf,
                                                 const int* __restrict__ eidx,
                                                 const float* __restrict__ stats1,
                                                 const float* __restrict__ eg1,
                                                 const float* __restrict__ eb1,
                                                 unsigned short* __restrict__ y2) {
  __shared__ unsigned short lA[128 * LDS_PITCH];
  __shared__ unsigned short lB[128 * LDS_PITCH];
  const int tid  = threadIdx.x;
  const int slot = blockIdx.z;
  const int mb   = blockIdx.y * 128;
  const int nb   = blockIdx.x * 128;
  const int e    = eidx[slot];
  const unsigned short* W = w2bf + (size_t)e * 256 * 512;
  const unsigned short* Xin = y1 + (size_t)slot * 512 * 4096;
  const int wave = tid >> 5, lane = tid & 31, l = lane & 15, hi = lane >> 4;
  v8f acc[8] = {};
  for (int k0 = 0; k0 < 512; k0 += 32) {
    #pragma unroll
    for (int j = 0; j < 4; ++j) {
      int idx = tid + j * 256;
      int m = idx >> 3, k4 = (idx & 7) * 4;
      uint2 pk = *(const uint2*)&W[(size_t)(mb + m) * 512 + k0 + k4];
      *(uint2*)&lA[m * LDS_PITCH + k4] = pk;
    }
    #pragma unroll
    for (int j = 0; j < 4; ++j) {
      int idx = tid + j * 256;
      int k = idx >> 5, n4 = (idx & 31) * 4;
      int kg = k0 + k;
      int g  = kg >> 6;                          // 64 channels / group
      float mean = stats1[(slot * 8 + g) * 2];
      float rs   = stats1[(slot * 8 + g) * 2 + 1];
      float sc   = rs * eg1[e * 512 + kg];
      float sb   = eb1[e * 512 + kg] - mean * sc;
      uint2 raw = *(const uint2*)&Xin[(size_t)kg * 4096 + nb + n4];
      float v0 = silu(bf2f((unsigned short)(raw.x & 0xFFFF)) * sc + sb);
      float v1 = silu(bf2f((unsigned short)(raw.x >> 16)) * sc + sb);
      float v2 = silu(bf2f((unsigned short)(raw.y & 0xFFFF)) * sc + sb);
      float v3 = silu(bf2f((unsigned short)(raw.y >> 16)) * sc + sb);
      lB[(n4 + 0) * LDS_PITCH + k] = f2bf(v0);
      lB[(n4 + 1) * LDS_PITCH + k] = f2bf(v1);
      lB[(n4 + 2) * LDS_PITCH + k] = f2bf(v2);
      lB[(n4 + 3) * LDS_PITCH + k] = f2bf(v3);
    }
    __syncthreads();
    const unsigned short* ar = &lA[(wave * 16 + l) * LDS_PITCH];
    v16bf a = frag16(ar + hi * 8, ar + 16 + hi * 8);
    #pragma unroll
    for (int t = 0; t < 8; ++t) {
      const unsigned short* br = &lB[(t * 16 + l) * LDS_PITCH];
      v16bf bb = frag16(br + hi * 8, br + 16 + hi * 8);
      acc[t] = __builtin_amdgcn_wmma_f32_16x16x32_bf16(
          false, a, false, bb, (short)0, acc[t], false, false);
    }
    __syncthreads();
  }
  unsigned short* Y = y2 + (size_t)slot * 256 * 4096;
  #pragma unroll
  for (int t = 0; t < 8; ++t)
    #pragma unroll
    for (int r = 0; r < 8; ++r) {
      int m = mb + wave * 16 + hi * 8 + r;
      int n = nb + t * 16 + l;
      Y[(size_t)m * 4096 + n] = f2bf(acc[t][r]);
    }
}

// ---------------------------------------------------------------------------
// Final: gn2 affine + routing weight + top-2 sum -> out[32,256,64,64] f32
// ---------------------------------------------------------------------------
__global__ __launch_bounds__(256) void moe_combine(const unsigned short* __restrict__ y2,
                                                   const float* __restrict__ stats2,
                                                   const int* __restrict__ eidx,
                                                   const float* __restrict__ wslot,
                                                   const float* __restrict__ eg2,
                                                   const float* __restrict__ eb2,
                                                   float* __restrict__ out) {
  size_t idx = (size_t)blockIdx.x * 256 + threadIdx.x;    // < 2^25
  int b = (int)(idx >> 20);
  int c = (int)((idx >> 12) & 255);
  int p = (int)(idx & 4095);
  float accv = 0.f;
  #pragma unroll
  for (int k = 0; k < 2; ++k) {
    int slot = b * 2 + k;
    int e = eidx[slot];
    int g = c >> 5;                                       // 32 ch / group
    float mean = stats2[(slot * 8 + g) * 2];
    float rs   = stats2[(slot * 8 + g) * 2 + 1];
    float v = bf2f(y2[((size_t)slot * 256 + c) * 4096 + p]);
    v = (v - mean) * rs * eg2[e * 256 + c] + eb2[e * 256 + c];
    accv += wslot[slot] * v;
  }
  out[idx] = accv;
}

// ---------------------------------------------------------------------------
extern "C" void kernel_launch(void* const* d_in, const int* in_sizes, int n_in,
                              void* d_out, int out_size, void* d_ws, size_t ws_size,
                              hipStream_t stream) {
  (void)in_sizes; (void)n_in; (void)out_size; (void)ws_size;
  const float* x    = (const float*)d_in[0];
  const float* rdw  = (const float*)d_in[1];
  const float* rg1  = (const float*)d_in[2];
  const float* rb1  = (const float*)d_in[3];
  const float* rpw1 = (const float*)d_in[4];
  const float* rg2  = (const float*)d_in[5];
  const float* rb2  = (const float*)d_in[6];
  const float* rpw2 = (const float*)d_in[7];
  const float* rpb2 = (const float*)d_in[8];
  const float* ew1  = (const float*)d_in[9];
  const float* eg1  = (const float*)d_in[10];
  const float* eb1  = (const float*)d_in[11];
  const float* ew2  = (const float*)d_in[12];
  const float* eg2  = (const float*)d_in[13];
  const float* eb2  = (const float*)d_in[14];
  float* out = (float*)d_out;

  // workspace layout
  float* xd     = (float*)d_ws;                  // 524288 f32
  float* hrt    = xd + 524288;                   // 524288 f32
  float* wslot  = hrt + 524288;                  // 64 f32
  int*   eidx   = (int*)(wslot + 64);            // 64 i32
  float* stats1 = (float*)(eidx + 64);           // 1024 f32
  float* stats2 = stats1 + 1024;                 // 1024 f32
  unsigned short* xbf  = (unsigned short*)(stats2 + 1024); // 33554432 bf16
  unsigned short* w1bf = xbf + (size_t)32 * 256 * 4096;    // 1048576 bf16
  unsigned short* w2bf = w1bf + (size_t)8 * 512 * 256;     // 1048576 bf16
  unsigned short* y1   = w2bf + (size_t)8 * 256 * 512;     // 64*512*4096 bf16
  unsigned short* y2   = y1 + (size_t)64 * 512 * 4096;     // 64*256*4096 bf16

  // operand pre-conversion to bf16
  cvt_bf16<<<32768, 256, 0, stream>>>(x, xbf, 8388608);
  cvt_bf16<<<1024, 256, 0, stream>>>(ew1, w1bf, 262144);
  cvt_bf16<<<1024, 256, 0, stream>>>(ew2, w2bf, 262144);

  // router
  r_pool<<<2048, 256, 0, stream>>>(x, xd);
  r_dwgn<<<32, 256, (16384 + 512) * sizeof(float), stream>>>(xd, rdw, rg1, rb1, hrt);
  r_route<<<32, 64, 17416 * sizeof(float), stream>>>(hrt, rpw1, rg2, rb2, rpw2, rpb2,
                                                     eidx, wslot);
  // experts
  moe_gemm1<<<dim3(32, 4, 64), 256, 0, stream>>>(xbf, w1bf, eidx, y1);
  moe_stats<<<dim3(8, 64), 256, 0, stream>>>(y1, stats1, 512, 64);
  moe_gemm2<<<dim3(32, 2, 64), 256, 0, stream>>>(y1, w2bf, eidx, stats1, eg1, eb1, y2);
  moe_stats<<<dim3(8, 64), 256, 0, stream>>>(y2, stats2, 256, 32);
  moe_combine<<<131072, 256, 0, stream>>>(y2, stats2, eidx, wslot, eg2, eb2, out);
}
